// TGCNN_layer_3607772529264
// MI455X (gfx1250) — compile-verified, matrix operand order checked by reading, not run on hardware
//
#include <hip/hip_runtime.h>

typedef __attribute__((ext_vector_type(16))) _Float16 v16h;
typedef __attribute__((ext_vector_type(8)))  _Float16 v8h;
typedef __attribute__((ext_vector_type(8)))  float    v8f;
typedef __attribute__((ext_vector_type(4)))  float    v4f;

#define B_     64
#define NN     10000          // N*N channels
#define T_     100
#define FS     3
#define F_     16
#define OUTT   98             // T - FS + 1
#define TT     16             // time tile (WMMA M)
#define NT     7              // tiles: t0 = 0,16,...,80,82 (last overlaps)
#define TL     (TT + FS - 1)  // 18 time samples per tile
#define CC     256            // channels per chunk (== blockDim)
#define NCHUNK 40             // ceil(NN / CC)
#define CPAD   (CC * NCHUNK)  // 10240 zero-padded channel dim for weights
#define NW     8              // waves per block (wave32)

// ---------------------------------------------------------------------------
// Pre-pass: w (30000x16 f32) -> f16 scratch, layout wh[dt][f][c_pad] so a
// WMMA B-fragment K-slice (32 channels, one dt) is two contiguous b128 loads
// per lane (lane<16: K={0..7,16..23}; lane>=16: K={8..15,24..31}).
// ---------------------------------------------------------------------------
__global__ __launch_bounds__(256) void convert_w_kernel(
    const float* __restrict__ w, _Float16* __restrict__ wh) {
  int idx = blockIdx.x * 256 + threadIdx.x;       // over FS*F_*CPAD
  if (idx >= FS * F_ * CPAD) return;
  int c  = idx % CPAD;
  int nf = idx / CPAD;                            // dt*16 + f
  int f  = nf % F_;
  int dt = nf / F_;
  float v = (c < NN) ? w[(c * FS + dt) * F_ + f] : 0.0f;
  wh[idx] = (_Float16)v;
}

// ---------------------------------------------------------------------------
// Branch-free chunk loop. OFF = t0 - ta (0 or 2), block-uniform, templated so
// the hot loop has zero control flow beyond the chunk counter. Double-buffered
// LDS tile -> one barrier per chunk.
// ---------------------------------------------------------------------------
template <int OFF>
__device__ __forceinline__ v8f tgcnn_chunks(
    const float* __restrict__ x, const _Float16* __restrict__ wh,
    _Float16 (&tb)[2][TL][CC], float k2,
    int b, int ta, int tid, int wave, int half, int l) {
  v8f acc = {};
  const int kb = wave * 32;

  for (int ch = 0; ch < NCHUNK; ++ch) {
    const int p = ch & 1;
    const int c  = ch * CC + tid;
    const int cl = (c < NN) ? c : (NN - 1);       // clamp: loads always valid
    const bool cok = (c < NN);

    // 20 floats, 16B-aligned (row base is 400B-aligned, ta multiple of 4)
    const float4* r4 =
        reinterpret_cast<const float4*>(x + ((size_t)b * NN + cl) * T_ + ta);
    float vals[20];
#pragma unroll
    for (int q = 0; q < 5; ++q) *(float4*)&vals[q * 4] = r4[q];

    // speculative prefetch of next chunk's rows (global_prefetch_b8)
    {
      const int cn  = ch * CC + CC + tid;
      const int cnc = (cn < NN) ? cn : (NN - 1);
      __builtin_prefetch(x + ((size_t)b * NN + cnc) * T_ + ta, 0, 1);
    }

    // xs = exp(-gamma*x) on nonzeros = exp2(k2*x);  f16 into LDS [t][c]
    _Float16 h[TL];
#pragma unroll
    for (int i = 0; i < TL; ++i) {
      float xv = vals[i + OFF];
      h[i] = (cok && xv != 0.0f)
                 ? (_Float16)__builtin_amdgcn_exp2f(k2 * xv)
                 : (_Float16)0.0f;
    }
#pragma unroll
    for (int i = 0; i < TL; ++i) tb[p][i][tid] = h[i];

    __syncthreads();

#pragma unroll
    for (int dt = 0; dt < FS; ++dt) {
      // A 16x32 f16: lane<16 -> M=l, K={0..7,16..23}; lane>=16 -> {8..15,24..31}
      v8h alo = *(const v8h*)&tb[p][l + dt][kb + half * 8];
      v8h ahi = *(const v8h*)&tb[p][l + dt][kb + 16 + half * 8];
      v16h a = __builtin_shufflevector(alo, ahi, 0, 1, 2, 3, 4, 5, 6, 7,
                                       8, 9, 10, 11, 12, 13, 14, 15);
      // B 32x16 f16: lane holds column N=l, same K grouping, contiguous in c
      const _Float16* wrow =
          wh + (size_t)(dt * F_ + l) * CPAD + ch * CC + kb;
      v8h blo = *(const v8h*)(wrow + half * 8);
      v8h bhi = *(const v8h*)(wrow + 16 + half * 8);
      v16h bm = __builtin_shufflevector(blo, bhi, 0, 1, 2, 3, 4, 5, 6, 7,
                                        8, 9, 10, 11, 12, 13, 14, 15);
      acc = __builtin_amdgcn_wmma_f32_16x16x32_f16(
          false, a, false, bm, (short)0, acc, false, false);
    }
  }
  return acc;
}

// ---------------------------------------------------------------------------
// Main kernel: block = (batch b, time tile). 8 waves; each wave owns a
// 32-channel K-slice per 256-channel chunk; partial 16x16 f32 accumulators
// reduced through LDS at the end.
// ---------------------------------------------------------------------------
__global__ __launch_bounds__(256) void tgcnn_wmma_kernel(
    const float*    __restrict__ x,       // (B, NN, T)
    const _Float16* __restrict__ wh,      // (FS, F, CPAD) f16
    const float*    __restrict__ gammat,  // (1,1)
    float*          __restrict__ out) {   // (B, F, 1, OUTT)
  __shared__ __align__(16) _Float16 tb[2][TL][CC];  // 2 x 18 x 256 f16 = 18 KB
  __shared__ __align__(16) float    red[NW][256];   // 8 KB partial-C exchange

  const int b    = blockIdx.y;
  const int tlid = blockIdx.x;
  const int t0   = (tlid == NT - 1) ? (T_ - TL) : tlid * TT;  // 82 for last
  const int ta   = t0 & ~3;                                   // aligned base
  const int tid  = threadIdx.x;
  const int wave = tid >> 5;
  const int lane = tid & 31;
  const int half = lane >> 4;
  const int l    = lane & 15;

  // gamma = 10*sigmoid(gammat); exp(-gamma*x) == exp2(k2*x)
  const float gamma = 10.0f / (1.0f + __expf(-gammat[0]));
  const float k2    = -gamma * 1.44269504088896340736f;

  v8f acc;
  if (t0 == ta)   // tiles 0..5 (uniform branch, taken once)
    acc = tgcnn_chunks<0>(x, wh, tb, k2, b, ta, tid, wave, half, l);
  else            // last tile, t0 = 82, OFF = 2
    acc = tgcnn_chunks<2>(x, wh, tb, k2, b, ta, tid, wave, half, l);

  // cross-wave reduction of 8 partial 16x16 accumulators
  v4f lo = __builtin_shufflevector(acc, acc, 0, 1, 2, 3);
  v4f hi = __builtin_shufflevector(acc, acc, 4, 5, 6, 7);
  *(v4f*)&red[wave][lane * 8 + 0] = lo;
  *(v4f*)&red[wave][lane * 8 + 4] = hi;
  __syncthreads();

  float s = 0.0f;
#pragma unroll
  for (int w = 0; w < NW; ++w) s += red[w][tid];

  // decode: tid = lane*8 + r ; C layout: VGPR r -> M = r + half*8, N = l
  const int r   = tid & 7;
  const int ln  = tid >> 3;
  const int hh  = ln >> 4;
  const int ll  = ln & 15;
  const int tl2 = r + hh * 8;        // time within tile
  const int f   = ll;                // filter
  const int t   = t0 + tl2;
  // each tile owns outputs [tlid*TT, min(tlid*TT+TT, OUTT)) -> no overlap
  if (t >= tlid * TT && t < OUTT)
    out[((size_t)b * F_ + f) * OUTT + t] = s;
}

extern "C" void kernel_launch(void* const* d_in, const int* in_sizes, int n_in,
                              void* d_out, int out_size, void* d_ws, size_t ws_size,
                              hipStream_t stream) {
  const float* x      = (const float*)d_in[0];
  const float* w      = (const float*)d_in[1];
  const float* gammat = (const float*)d_in[2];
  float* out          = (float*)d_out;
  _Float16* wh        = (_Float16*)d_ws;   // FS*F_*CPAD f16 = 960 KB

  const int wtot = FS * F_ * CPAD;
  convert_w_kernel<<<(wtot + 255) / 256, 256, 0, stream>>>(w, wh);
  tgcnn_wmma_kernel<<<dim3(NT, B_), 256, 0, stream>>>(x, wh, gammat, out);
}
